// SMFE_68753836474921
// MI455X (gfx1250) — matrix operand order repaired and checked by reading, not compiled, not run on hardware
//
#include <hip/hip_runtime.h>
#include <hip/hip_bf16.h>

typedef __attribute__((ext_vector_type(16))) _Float16 v16h;
typedef __attribute__((ext_vector_type(4)))  _Float16 v4h;
typedef __attribute__((ext_vector_type(8)))  float    v8f;

#define WMMA16(a, b, c) \
    __builtin_amdgcn_wmma_f32_16x16x32_f16(false, (a), false, (b), (short)0, (c), false, false)

namespace {
constexpr int kB = 8, kS = 1024, kD = 512, kH = 8, kHD = 64;
constexpr float kNEG = -1.0e9f;
}

union AFrag {
    v16h h;
    uint4 u[2];
};

// ---------------------------------------------------------------------------
// fp32 -> fp16 cast
// ---------------------------------------------------------------------------
__global__ __launch_bounds__(256) void cast_f16_kernel(const float* __restrict__ in,
                                                       _Float16* __restrict__ out, int n) {
    int i = blockIdx.x * 256 + threadIdx.x;
    if (i < n) out[i] = (_Float16)in[i];
}

// ---------------------------------------------------------------------------
// Generic 8192x512x512 GEMM: C = A(M,512) * W(512,512)^T + bias
// W is stored row-major [n][k] (== B(k,n) column access pattern for WMMA B).
// One wave per block; each wave computes a 32x64 output strip (2 M sub-tiles
// sharing every B fragment -> ~2x arithmetic intensity vs 16x64).
// mode 0: scatter f16 to [b][h][s][hd]     (Q, K)
// mode 3: scatter f16 to [b][h][hd][s]     (V transposed)
// mode 1: write f32 + f16 copies, row-major [m][n]   (Wo projection)
// mode 2: write f32 = resid + acc + bias   (W1 + residual, pre-LN)
// ---------------------------------------------------------------------------
__global__ __launch_bounds__(32) void gemm512_wmma_kernel(
    const _Float16* __restrict__ A, const _Float16* __restrict__ W,
    const float* __restrict__ bias, _Float16* __restrict__ o16, float* __restrict__ o32,
    const float* __restrict__ resid, int mode) {
    const int L = threadIdx.x;
    const int half = L >> 4;
    const int lc = L & 15;
    const int N0 = blockIdx.x * 64;
    const int M0 = blockIdx.y * 32;

    v8f zero = {};
    v8f acc[2][4] = {{zero, zero, zero, zero}, {zero, zero, zero, zero}};

    const _Float16* arow0 = A + (size_t)(M0 + lc) * kD;       // M sub-tile 0
    const _Float16* arow1 = A + (size_t)(M0 + 16 + lc) * kD;  // M sub-tile 1
    const int alo = half * 8;                                 // A K-chunk offset per lane half
    const int blo = half * 8;                                 // B K-chunk offset per lane half

    for (int k0 = 0; k0 < kD; k0 += 32) {
        AFrag a0, a1;
        a0.u[0] = *(const uint4*)(arow0 + k0 + alo);
        a0.u[1] = *(const uint4*)(arow0 + k0 + 16 + alo);
        a1.u[0] = *(const uint4*)(arow1 + k0 + alo);
        a1.u[1] = *(const uint4*)(arow1 + k0 + 16 + alo);
        if (k0 + 32 < kD) {
            __builtin_prefetch(arow0 + k0 + 32 + alo, 0, 1);
            __builtin_prefetch(arow1 + k0 + 32 + alo, 0, 1);
        }
#pragma unroll
        for (int j = 0; j < 4; j++) {
            const _Float16* wrow = W + (size_t)(N0 + 16 * j + lc) * kD + k0 + blo;
            AFrag bu;
            bu.u[0] = *(const uint4*)(wrow);
            bu.u[1] = *(const uint4*)(wrow + 16);
            v16h bf = bu.h;
            acc[0][j] = WMMA16(a0.h, bf, acc[0][j]);
            acc[1][j] = WMMA16(a1.h, bf, acc[1][j]);
        }
    }

#pragma unroll
    for (int mt = 0; mt < 2; mt++) {
#pragma unroll
        for (int j = 0; j < 4; j++) {
            const int n = N0 + 16 * j + lc;
            const float bn = bias ? bias[n] : 0.0f;
#pragma unroll
            for (int r = 0; r < 8; r++) {
                const int m = M0 + mt * 16 + r + half * 8;  // C layout: M = r + 8*(lane>=16)
                const float v = acc[mt][j][r] + bn;
                if (mode == 0) {
                    const int b = m >> 10, s = m & 1023, h = n >> 6, hd = n & 63;
                    o16[((size_t)((b * kH + h) * kS + s)) * kHD + hd] = (_Float16)v;
                } else if (mode == 3) {
                    const int b = m >> 10, s = m & 1023, h = n >> 6, hd = n & 63;
                    o16[((size_t)((b * kH + h) * kHD + hd)) * kS + s] = (_Float16)v;
                } else if (mode == 1) {
                    const size_t off = (size_t)m * kD + n;
                    o32[off] = v;
                    o16[off] = (_Float16)v;
                } else {  // mode 2
                    const size_t off = (size_t)m * kD + n;
                    o32[off] = v + resid[off];
                }
            }
        }
    }
}

// ---------------------------------------------------------------------------
// Memory-attention softmax (head-independent, computed ONCE per (b,q) row):
// P_mem[b][q][k] = softmax_k( masked(log10(ren+1) + exp(-|ts|)) ), f16 out.
// One 256-thread block per row, 4 keys per thread.
// ---------------------------------------------------------------------------
__global__ __launch_bounds__(256) void mem_softmax_kernel(const float* __restrict__ ren,
                                                          const float* __restrict__ ts,
                                                          _Float16* __restrict__ P) {
    const int row = blockIdx.x;  // b*1024 + q
    const int q = row & 1023;
    const size_t base = (size_t)row * kS;
    const int t = threadIdx.x;

    float f[4];
#pragma unroll
    for (int i = 0; i < 4; i++) {
        const int key = t * 4 + i;
        if (key <= q) {
            f[i] = log10f(ren[base + key] + 1.0f) + expf(-fabsf(ts[base + key]));
        } else {
            f[i] = 2.0f * kNEG;  // both paths masked to NEG then summed in reference
        }
    }
    float mx = fmaxf(fmaxf(f[0], f[1]), fmaxf(f[2], f[3]));
    for (int o = 1; o < 32; o <<= 1) mx = fmaxf(mx, __shfl_xor(mx, o));

    __shared__ float red[8];
    if ((t & 31) == 0) red[t >> 5] = mx;
    __syncthreads();
    float gmax = red[0];
#pragma unroll
    for (int w = 1; w < 8; w++) gmax = fmaxf(gmax, red[w]);
    __syncthreads();

    float e[4], s = 0.0f;
#pragma unroll
    for (int i = 0; i < 4; i++) {
        e[i] = expf(f[i] - gmax);
        s += e[i];
    }
    for (int o = 1; o < 32; o <<= 1) s += __shfl_xor(s, o);
    if ((t & 31) == 0) red[t >> 5] = s;
    __syncthreads();
    float gsum = 0.0f;
#pragma unroll
    for (int w = 0; w < 8; w++) gsum += red[w];
    const float inv = 1.0f / gsum;

    v4h pv;
#pragma unroll
    for (int i = 0; i < 4; i++) pv[i] = (_Float16)(e[i] * inv);
    *(v4h*)(P + base + t * 4) = pv;
}

// ---------------------------------------------------------------------------
// Flash attention: one wave per (b, h, 16-row q-tile).
// scores = (q @ k^T)/8 with causal mask -> online softmax -> P @ V   (accP)
// P_mem (precomputed, normalized)        ->                 P_mem @ V (accM)
// out = (1-l)*accP/rowsum + l*accM, written f16 to [b][s][d].
// Causal tiles with key_base > q_max are skipped entirely.
// ---------------------------------------------------------------------------
__global__ __launch_bounds__(32) void attn_wmma_kernel(
    const _Float16* __restrict__ q16, const _Float16* __restrict__ k16,
    const _Float16* __restrict__ vT, const _Float16* __restrict__ Pmem,
    const float* __restrict__ l1, _Float16* __restrict__ out) {
    const int L = threadIdx.x;
    const int half = L >> 4;
    const int lc = L & 15;
    const int blk = blockIdx.x;
    const int qt = blk & 63;
    const int bh = blk >> 6;  // b*H + h
    const int b = bh >> 3;
    const int h = bh & 7;
    const int q0 = qt * 16;

    __shared__ __align__(16) _Float16 ldsP[16 * 32];  // prob tile, C-layout -> A-layout bounce

    const int alo = half * 8;
    const _Float16* qrow = q16 + ((size_t)(bh * kS + q0 + lc)) * kHD;
    AFrag qa0, qa1;
    qa0.u[0] = *(const uint4*)(qrow + alo);
    qa0.u[1] = *(const uint4*)(qrow + 16 + alo);
    qa1.u[0] = *(const uint4*)(qrow + 32 + alo);
    qa1.u[1] = *(const uint4*)(qrow + 48 + alo);

    float rmax[8], rsum[8];
#pragma unroll
    for (int r = 0; r < 8; r++) {
        rmax[r] = -1.0e30f;
        rsum[r] = 0.0f;
    }
    v8f zero = {};
    v8f accP[4] = {zero, zero, zero, zero};
    v8f accM[4] = {zero, zero, zero, zero};

    const _Float16* pmrow = Pmem + ((size_t)(b * kS + q0 + lc)) * kS;
    const int nkb = (q0 + 16 + 31) >> 5;  // causal: only key blocks with key <= q_max

    for (int kb = 0; kb < nkb; kb++) {
        const int kbase = kb * 32;
        // ---- two 16x16 score sub-tiles (keys kbase..kbase+31) ----
        v8f sc[2];
#pragma unroll
        for (int t = 0; t < 2; t++) {
            const int key = kbase + 16 * t + lc;
            const _Float16* kp = k16 + ((size_t)(bh * kS + key)) * kHD + half * 8;
            AFrag k0f, k1f;
            k0f.u[0] = *(const uint4*)(kp);
            k0f.u[1] = *(const uint4*)(kp + 16);
            k1f.u[0] = *(const uint4*)(kp + 32);
            k1f.u[1] = *(const uint4*)(kp + 48);
            v16h kb0 = k0f.h;  // hd chunk 0..31
            v16h kb1 = k1f.h;  // hd chunk 32..63
            v8f s4 = {};
            s4 = WMMA16(qa0.h, kb0, s4);
            s4 = WMMA16(qa1.h, kb1, s4);
            sc[t] = s4;
        }
        // ---- mask + scale + joint row-max over both sub-tiles ----
        const int key0 = kbase + lc;
        const int key1 = kbase + 16 + lc;
        float nmax[8];
#pragma unroll
        for (int r = 0; r < 8; r++) {
            const int m = q0 + r + half * 8;
            const float s0 = (key0 <= m) ? sc[0][r] * 0.125f : kNEG;
            const float s1 = (key1 <= m) ? sc[1][r] * 0.125f : kNEG;
            sc[0][r] = s0;
            sc[1][r] = s1;
            float tm = fmaxf(s0, s1);
            tm = fmaxf(tm, __shfl_xor(tm, 1));
            tm = fmaxf(tm, __shfl_xor(tm, 2));
            tm = fmaxf(tm, __shfl_xor(tm, 4));
            tm = fmaxf(tm, __shfl_xor(tm, 8));
            nmax[r] = fmaxf(rmax[r], tm);
        }
        // ---- exp, LDS transpose store, row sums, accumulator rescale ----
#pragma unroll
        for (int r = 0; r < 8; r++) {
            const float corr = expf(rmax[r] - nmax[r]);
            const float p0 = expf(sc[0][r] - nmax[r]);
            const float p1 = expf(sc[1][r] - nmax[r]);
            const int ml = r + half * 8;
            ldsP[ml * 32 + lc] = (_Float16)p0;
            ldsP[ml * 32 + 16 + lc] = (_Float16)p1;
            float ps = p0 + p1;
            ps += __shfl_xor(ps, 1);
            ps += __shfl_xor(ps, 2);
            ps += __shfl_xor(ps, 4);
            ps += __shfl_xor(ps, 8);
            rsum[r] = rsum[r] * corr + ps;
            rmax[r] = nmax[r];
#pragma unroll
            for (int j = 0; j < 4; j++) accP[j][r] *= corr;
        }
        __syncthreads();  // single-wave WG: barrier degrades to waitcnt/s_nop
        // ---- prob P tile as A-fragment (from LDS), mem P tile (from global) ----
        AFrag pa, ma;
        pa.u[0] = *(const uint4*)(&ldsP[lc * 32 + alo]);
        pa.u[1] = *(const uint4*)(&ldsP[lc * 32 + 16 + alo]);
        const _Float16* pm = pmrow + kbase;
        ma.u[0] = *(const uint4*)(pm + alo);
        ma.u[1] = *(const uint4*)(pm + 16 + alo);
        // ---- P @ V and P_mem @ V, sharing V B-fragments (vT is [b][h][hd][s]) ----
#pragma unroll
        for (int j = 0; j < 4; j++) {
            const _Float16* vp = vT + (size_t)(bh * kHD + 16 * j + lc) * kS + kbase + half * 8;
            AFrag vu;
            vu.u[0] = *(const uint4*)(vp);
            vu.u[1] = *(const uint4*)(vp + 16);
            v16h vf = vu.h;
            accP[j] = WMMA16(pa.h, vf, accP[j]);
            accM[j] = WMMA16(ma.h, vf, accM[j]);
        }
        __syncthreads();
    }

    const float l = l1[0];
    const float oml = 1.0f - l;
#pragma unroll
    for (int j = 0; j < 4; j++) {
#pragma unroll
        for (int r = 0; r < 8; r++) {
            const int m = q0 + r + half * 8;
            const float v = oml * accP[j][r] / rsum[r] + l * accM[j][r];
            out[((size_t)(b * kS + m)) * kD + h * kHD + 16 * j + lc] = (_Float16)v;
        }
    }
}

// ---------------------------------------------------------------------------
// In-place LayerNorm over 512-wide rows of fp32.
// ---------------------------------------------------------------------------
__global__ __launch_bounds__(256) void layernorm_kernel(float* __restrict__ y,
                                                        const float* __restrict__ g,
                                                        const float* __restrict__ bt) {
    const int row = blockIdx.x;
    float* p = y + (size_t)row * kD;
    const int t = threadIdx.x;
    const float x0 = p[t];
    const float x1 = p[t + 256];
    float s = x0 + x1;
    float ss = x0 * x0 + x1 * x1;
    for (int o = 1; o < 32; o <<= 1) {
        s += __shfl_xor(s, o);
        ss += __shfl_xor(ss, o);
    }
    __shared__ float rs[8], rss[8];
    if ((t & 31) == 0) {
        rs[t >> 5] = s;
        rss[t >> 5] = ss;
    }
    __syncthreads();
    float S1 = 0.0f, S2 = 0.0f;
#pragma unroll
    for (int w = 0; w < 8; w++) {
        S1 += rs[w];
        S2 += rss[w];
    }
    const float mu = S1 * (1.0f / 512.0f);
    const float var = S2 * (1.0f / 512.0f) - mu * mu;
    const float inv = rsqrtf(var + 1e-5f);
    p[t] = (x0 - mu) * inv * g[t] + bt[t];
    p[t + 256] = (x1 - mu) * inv * g[t + 256] + bt[t + 256];
}

// ---------------------------------------------------------------------------
extern "C" void kernel_launch(void* const* d_in, const int* in_sizes, int n_in, void* d_out,
                              int out_size, void* d_ws, size_t ws_size, hipStream_t stream) {
    (void)in_sizes;
    (void)n_in;
    (void)out_size;
    (void)ws_size;
    const float* f_x = (const float*)d_in[1];  // 'inputs' (note: 'query' d_in[0] unused by ref)
    const float* f_ren = (const float*)d_in[2];
    const float* f_ts = (const float*)d_in[3];
    const float* f_Wq = (const float*)d_in[5];
    const float* f_bq = (const float*)d_in[6];
    const float* f_Wk = (const float*)d_in[7];
    const float* f_bk = (const float*)d_in[8];
    const float* f_Wv = (const float*)d_in[9];
    const float* f_bv = (const float*)d_in[10];
    const float* f_Wo = (const float*)d_in[11];
    const float* f_bo = (const float*)d_in[12];
    const float* f_W1 = (const float*)d_in[13];
    const float* f_b1 = (const float*)d_in[14];
    const float* f_g = (const float*)d_in[15];
    const float* f_bt = (const float*)d_in[16];
    const float* f_l1 = (const float*)d_in[17];
    float* out = (float*)d_out;

    char* ws = (char*)d_ws;
    size_t off = 0;
    auto take = [&](size_t bytes) -> char* {
        char* p = ws + off;
        off += (bytes + 255) & ~(size_t)255;
        return p;
    };
    const size_t nBSD = (size_t)kB * kS * kD;  // 4,194,304
    const size_t nDD = (size_t)kD * kD;        // 262,144
    const size_t nBSS = (size_t)kB * kS * kS;  // 8,388,608

    _Float16* x16 = (_Float16*)take(nBSD * 2);
    _Float16* Wq16 = (_Float16*)take(nDD * 2);
    _Float16* Wk16 = (_Float16*)take(nDD * 2);
    _Float16* Wv16 = (_Float16*)take(nDD * 2);
    _Float16* Wo16 = (_Float16*)take(nDD * 2);
    _Float16* W116 = (_Float16*)take(nDD * 2);
    _Float16* q16 = (_Float16*)take(nBSD * 2);
    _Float16* k16 = (_Float16*)take(nBSD * 2);
    _Float16* vT16 = (_Float16*)take(nBSD * 2);
    _Float16* Pmem = (_Float16*)take(nBSS * 2);
    _Float16* attn16 = (_Float16*)take(nBSD * 2);
    _Float16* proj16 = (_Float16*)take(nBSD * 2);
    float* proj32 = (float*)take(nBSD * 4);

    // 1) fp16 casts
    cast_f16_kernel<<<(int)((nBSD + 255) / 256), 256, 0, stream>>>(f_x, x16, (int)nBSD);
    cast_f16_kernel<<<(int)((nDD + 255) / 256), 256, 0, stream>>>(f_Wq, Wq16, (int)nDD);
    cast_f16_kernel<<<(int)((nDD + 255) / 256), 256, 0, stream>>>(f_Wk, Wk16, (int)nDD);
    cast_f16_kernel<<<(int)((nDD + 255) / 256), 256, 0, stream>>>(f_Wv, Wv16, (int)nDD);
    cast_f16_kernel<<<(int)((nDD + 255) / 256), 256, 0, stream>>>(f_Wo, Wo16, (int)nDD);
    cast_f16_kernel<<<(int)((nDD + 255) / 256), 256, 0, stream>>>(f_W1, W116, (int)nDD);

    // 2) Q/K/V projections (WMMA GEMMs); V written transposed for P@V B-operand
    dim3 ggrid(kD / 64, (kB * kS) / 32);
    gemm512_wmma_kernel<<<ggrid, 32, 0, stream>>>(x16, Wq16, f_bq, q16, nullptr, nullptr, 0);
    gemm512_wmma_kernel<<<ggrid, 32, 0, stream>>>(x16, Wk16, f_bk, k16, nullptr, nullptr, 0);
    gemm512_wmma_kernel<<<ggrid, 32, 0, stream>>>(x16, Wv16, f_bv, vT16, nullptr, nullptr, 3);

    // 3) head-independent memory-attention softmax (64MB stream read exactly once)
    mem_softmax_kernel<<<kB * kS, 256, 0, stream>>>(f_ren, f_ts, Pmem);

    // 4) fused dual-path flash attention
    attn_wmma_kernel<<<kB * kH * (kS / 16), 32, 0, stream>>>(q16, k16, vT16, Pmem, f_l1, attn16);

    // 5) output projection (keeps f32 for residual + f16 for next GEMM)
    gemm512_wmma_kernel<<<ggrid, 32, 0, stream>>>(attn16, Wo16, f_bo, proj16, proj32, nullptr, 1);

    // 6) W1 GEMM + residual add, written straight into d_out (fp32)
    gemm512_wmma_kernel<<<ggrid, 32, 0, stream>>>(proj16, W116, f_b1, nullptr, out, proj32, 2);

    // 7) in-place LayerNorm
    layernorm_kernel<<<kB * kS, 256, 0, stream>>>(out, f_g, f_bt);
}
